// Attention_41996190220419
// MI455X (gfx1250) — compile-verified, matrix operand order checked by reading, not run on hardware
//
#include <hip/hip_runtime.h>
#include <hip/hip_bf16.h>
#include <math.h>

typedef __attribute__((ext_vector_type(16))) __bf16 v16bf;
typedef __attribute__((ext_vector_type(8)))  __bf16 v8bf;
typedef __attribute__((ext_vector_type(8)))  float  v8f;

#define DEV __device__ __forceinline__

constexpr int Bb = 8, Tt = 2048, Cc = 1024;
constexpr int BT = Bb * Tt;

// ---------- bf16 helpers (round-to-nearest-even) ----------
DEV __bf16 f2bf(float x) {
    union { float f; unsigned u; } in; in.f = x;
    unsigned u = in.u + 0x7FFFu + ((in.u >> 16) & 1u);
    union { unsigned short s; __bf16 b; } out; out.s = (unsigned short)(u >> 16);
    return out.b;
}
DEV float bf2f(__bf16 b) {
    union { unsigned short s; __bf16 b; } in; in.b = b;
    union { unsigned u; float f; } out; out.u = ((unsigned)in.s) << 16;
    return out.f;
}
DEV float softplusf(float x) { return x > 20.f ? x : log1pf(expf(x)); }

// Load one 16-element-per-lane WMMA fragment: two contiguous 8x bf16 (b128)
// chunks at +0 and +16 elements.  `p` must already include the per-lane
// (lane>=16 ? +8 : +0) K offset; 16-byte aligned.  Works for global or LDS.
DEV v16bf load_frag(const __bf16* p) {
    v8bf lo = *(const v8bf*)p;
    v8bf hi = *(const v8bf*)(p + 16);
    v16bf r;
#pragma unroll
    for (int i = 0; i < 8; ++i) { r[i] = lo[i]; r[i + 8] = hi[i]; }
    return r;
}

DEV v8f wmma_bf16(v16bf a, v16bf b, v8f c) {
    return __builtin_amdgcn_wmma_f32_16x16x32_bf16(false, a, false, b,
                                                   (short)0, c, false, false);
}

// Async 16-byte copy global -> LDS (tracked by ASYNCcnt).
// lds_byte_off: byte offset within the kernel's LDS allocation (static
// __shared__ arrays start at offset 0 when they are the only static object).
// lds_ptr: generic pointer to the same destination -- passed into the asm so
// the shared array's address ESCAPES and alias analysis must assume the asm
// writes it (otherwise loads from the never-visibly-stored LDS fold to undef,
// which is exactly what round 2's codegen showed).
DEV void async_copy_b128(unsigned lds_byte_off, void* lds_ptr, const void* gptr) {
    asm volatile("global_load_async_to_lds_b128 %0, %1, off"
                 :: "v"(lds_byte_off), "v"((unsigned long long)gptr), "v"(lds_ptr)
                 : "memory");
}
DEV void wait_async0() {
    asm volatile("s_wait_asynccnt 0x0" ::: "memory");
}

// ---------- elementwise / layout kernels ----------
__global__ void cvt_f32_bf16(const float* __restrict__ in, __bf16* __restrict__ out, int n) {
    int i = blockIdx.x * blockDim.x + threadIdx.x;
    if (i < n) out[i] = f2bf(in[i]);
}

// W (K x N, f32, row-major) -> Wt (N x K, bf16)
__global__ void transpose_cvt_w(const float* __restrict__ W, __bf16* __restrict__ Wt,
                                int K, int N) {
    int i = blockIdx.x * blockDim.x + threadIdx.x;
    if (i < K * N) {
        int k = i / N, n = i % N;
        Wt[(size_t)n * K + k] = f2bf(W[i]);
    }
}

// (B,T,C) bf16 -> (B,C,T) bf16
__global__ void transpose_v(const __bf16* __restrict__ V, __bf16* __restrict__ Vt, int n) {
    int i = blockIdx.x * blockDim.x + threadIdx.x;
    if (i < n) {
        int c = i % Cc, t = (i / Cc) % Tt, b = i / (Cc * Tt);
        Vt[(size_t)b * Cc * Tt + (size_t)c * Tt + t] = V[i];
    }
}

// In-place pairwise rotation, angle[t][j] = (T-1-t)*j  (flipped outer product)
__global__ void rope_rotate(__bf16* __restrict__ Z, int n) {
    int i = blockIdx.x * blockDim.x + threadIdx.x;   // over B*T*(C/2)
    if (i >= n) return;
    const int halfc = Cc / 2;
    int j = i % halfc;
    int t = (i / halfc) % Tt;
    size_t base = (size_t)(i / halfc) * Cc + 2 * j;
    float ang = (float)(Tt - 1 - t) * (float)j;
    float s, c;
    sincosf(ang, &s, &c);
    float z0 = bf2f(Z[base]), z1 = bf2f(Z[base + 1]);
    Z[base]     = f2bf(c * z0 + s * z1);
    Z[base + 1] = f2bf(-s * z0 + c * z1);
}

// ---------- GEMM: out = A(MxK,bf16) * Bt(NxK,bf16)^T + bias ----------
// One wave -> 32x64 output tile (2x4 WMMA accumulators). blockDim = 128.
// The 32-row A panel is shared by all 4 waves: staged cooperatively into LDS
// with double-buffered async-to-LDS copies overlapping the WMMAs.
__global__ void __launch_bounds__(128)
gemm_bf16(const __bf16* __restrict__ A, const __bf16* __restrict__ Bt,
          const float* __restrict__ bias,
          __bf16* __restrict__ outb, float* __restrict__ outf,
          int M, int N, int K) {
    __shared__ __bf16 Atile[2][32 * 32];            // 2 x 2KB double buffer
    const int tid  = threadIdx.x;
    const int lane = threadIdx.x & 31;
    const int wave = threadIdx.x >> 5;
    const int mbase = blockIdx.x * 32;
    const int nbase = blockIdx.y * 256 + wave * 64;
    const int r16 = lane & 15;
    const int half = (lane >= 16) ? 8 : 0;

    // stage A[mbase..mbase+31][k0..k0+31] into Atile[buf]; 128 chunks of 16B,
    // one per thread: chunk c=tid -> row c/4, elems (c%4)*8..+7 -> flat c*8
    auto stage = [&](int buf, int k0) {
        const int row = tid >> 2, kc = tid & 3;
        async_copy_b128((unsigned)(buf * 2048 + tid * 16),
                        &Atile[buf][tid * 8],
                        A + (size_t)(mbase + row) * K + k0 + kc * 8);
    };

    v8f acc[2][4];
#pragma unroll
    for (int mi = 0; mi < 2; ++mi)
#pragma unroll
        for (int ni = 0; ni < 4; ++ni)
            acc[mi][ni] = (v8f){0.f, 0.f, 0.f, 0.f, 0.f, 0.f, 0.f, 0.f};

    const __bf16* bp0 = Bt + (size_t)(nbase + r16) * K + half;

    stage(0, 0);
    for (int k0 = 0; k0 < K; k0 += 32) {
        const int cur = (k0 >> 5) & 1;
        wait_async0();                               // my writes to Atile[cur] done
        __syncthreads();                             // everyone's writes done
        if (k0 + 32 < K) stage(cur ^ 1, k0 + 32);    // overlap next panel copy

        __builtin_prefetch(bp0 + k0 + 256, 0, 1);    // -> global_prefetch_b8
        const __bf16* Ab = &Atile[cur][0];
        v16bf a0 = load_frag(Ab + r16 * 32 + half);          // ds_load_b128 x2
        v16bf a1 = load_frag(Ab + (r16 + 16) * 32 + half);
        v16bf b[4];
#pragma unroll
        for (int ni = 0; ni < 4; ++ni)
            b[ni] = load_frag(bp0 + (size_t)ni * 16 * K + k0);
#pragma unroll
        for (int ni = 0; ni < 4; ++ni) {
            acc[0][ni] = wmma_bf16(a0, b[ni], acc[0][ni]);
            acc[1][ni] = wmma_bf16(a1, b[ni], acc[1][ni]);
        }
        __syncthreads();                             // all reads of cur done before
    }                                                // it is restaged next iter

#pragma unroll
    for (int mi = 0; mi < 2; ++mi)
#pragma unroll
        for (int ni = 0; ni < 4; ++ni)
#pragma unroll
            for (int v = 0; v < 8; ++v) {
                int row = mbase + mi * 16 + v + half;   // C-layout: M = v + 8*(lane>=16)
                int col = nbase + ni * 16 + r16;        // N = lane%16
                float val = acc[mi][ni][v] + (bias ? bias[col] : 0.f);
                if (outb) outb[(size_t)row * N + col] = f2bf(val);
                else      outf[(size_t)row * N + col] = val;
            }
}

// ---------- attention: out[m,n] = sum_j softplus(scale*K_m.Q_j) V[j,n], j<=m ----
// grid = (T/16, B), block = 256 (8 waves), dyn LDS = 16*T bf16 (64 KB) for P.
// The 16x1024 K-tile (B-operand of every phase-1 WMMA) is staged once into a
// 32 KB static LDS buffer via async copies; total LDS 96 KB < 320 KB WGP pool.
__global__ void __launch_bounds__(256)
attn_kernel(const __bf16* __restrict__ Kr, const __bf16* __restrict__ Qr,
            const __bf16* __restrict__ Vt, __bf16* __restrict__ Ao) {
    __shared__ __bf16 Ktile[16 * Cc];             // 32 KB, LDS offset 0
    extern __shared__ __bf16 P[];                 // P[m_local][j], stride Tt
    const int tid  = threadIdx.x;
    const int lane = threadIdx.x & 31;
    const int wave = threadIdx.x >> 5;
    const int b = blockIdx.y;
    const int mbase = blockIdx.x * 16;
    const int r16 = lane & 15;
    const int half = (lane >= 16) ? 8 : 0;
    const int numJ = (mbase + 16 + 31) >> 5;      // 32-wide j tiles up to diagonal
    const float scale = 0.03125f;                 // 1/sqrt(C)

    const __bf16* Kb = Kr + (size_t)b * Tt * Cc;
    const __bf16* Qb = Qr + (size_t)b * Tt * Cc;

    // Stage K-tile: 2048 chunks of 16B across 256 threads (8 each);
    // chunk c -> row c/128, elems (c%128)*8 -> flat c*8.
#pragma unroll
    for (int s = 0; s < 8; ++s) {
        int c = tid + s * 256;
        int row = c >> 7, kc = c & 127;
        async_copy_b128((unsigned)(c * 16), &Ktile[c * 8],
                        Kb + (size_t)(mbase + row) * Cc + kc * 8);
    }
    wait_async0();
    __syncthreads();

    // Phase 1: S'[j][m] = Q_j . K_m  (= wei[m][j]); mask+softplus -> P[m][j]
    const __bf16* kp = &Ktile[0] + r16 * Cc + half;          // B-frags from LDS
    for (int jt = wave; jt < numJ; jt += 8) {
        const int jbase = jt * 32;
        v8f Sa = (v8f){0.f,0.f,0.f,0.f,0.f,0.f,0.f,0.f};
        v8f Sb = Sa;
        const __bf16* qa0 = Qb + (size_t)(jbase + r16) * Cc + half;       // A: rows j..j+15
        const __bf16* qa1 = Qb + (size_t)(jbase + 16 + r16) * Cc + half;  // A: rows j+16..j+31
        for (int c0 = 0; c0 < Cc; c0 += 32) {
            v16bf kf = load_frag(kp + c0);                   // ds_load_b128 x2
            v16bf a0 = load_frag(qa0 + c0);
            v16bf a1 = load_frag(qa1 + c0);
            Sa = wmma_bf16(a0, kf, Sa);
            Sb = wmma_bf16(a1, kf, Sb);
        }
        const int m = mbase + r16;                // this lane's wei-row (column of S')
#pragma unroll
        for (int v = 0; v < 8; ++v) {
            int jA = jbase + v + half;            // C-layout row index
            int jB = jA + 16;
            float fa = (jA <= m) ? softplusf(Sa[v] * scale) : 0.f;
            float fb = (jB <= m) ? softplusf(Sb[v] * scale) : 0.f;
            P[r16 * Tt + jA] = f2bf(fa);
            P[r16 * Tt + jB] = f2bf(fb);
        }
    }
    __syncthreads();

    // Phase 2: O[m][n] += P[m][j] * Vt[n][j]; each wave owns 128 n-columns
    const __bf16* Vb = Vt + (size_t)b * Cc * Tt;
    const int nbase = wave * 128;
    v8f acc[8];
#pragma unroll
    for (int ni = 0; ni < 8; ++ni)
        acc[ni] = (v8f){0.f,0.f,0.f,0.f,0.f,0.f,0.f,0.f};

    for (int jt = 0; jt < numJ; ++jt) {
        const int jbase = jt * 32;
        v16bf a = load_frag(P + r16 * Tt + jbase + half);    // ds_load_b128 x2
#pragma unroll
        for (int ni = 0; ni < 8; ++ni) {
            v16bf bf = load_frag(Vb + (size_t)(nbase + ni * 16 + r16) * Tt + jbase + half);
            acc[ni] = wmma_bf16(a, bf, acc[ni]);
        }
    }

    __bf16* Aob = Ao + (size_t)b * Tt * Cc;
#pragma unroll
    for (int ni = 0; ni < 8; ++ni)
#pragma unroll
        for (int v = 0; v < 8; ++v) {
            int row = mbase + v + half;
            int col = nbase + ni * 16 + r16;
            Aob[(size_t)row * Cc + col] = f2bf(acc[ni][v]);
        }
}

// ---------- host launcher ----------
extern "C" void kernel_launch(void* const* d_in, const int* in_sizes, int n_in,
                              void* d_out, int out_size, void* d_ws, size_t ws_size,
                              hipStream_t stream) {
    (void)in_sizes; (void)n_in; (void)out_size; (void)ws_size;
    const float* x  = (const float*)d_in[0];
    const float* Wk = (const float*)d_in[1]; const float* bk = (const float*)d_in[2];
    const float* Wq = (const float*)d_in[3]; const float* bq = (const float*)d_in[4];
    const float* Wv = (const float*)d_in[5]; const float* bv = (const float*)d_in[6];
    const float* Wp = (const float*)d_in[7]; const float* bp = (const float*)d_in[8];
    float* out = (float*)d_out;

    char* w = (char*)d_ws;
    auto alloc = [&](size_t bytes) -> char* {
        char* p = w; w += (bytes + 255) & ~(size_t)255; return p;
    };
    const size_t actBytes = (size_t)BT * Cc * 2;
    const size_t wBytes   = (size_t)Cc * Cc * 2;
    __bf16* xb  = (__bf16*)alloc(actBytes);
    __bf16* Wkt = (__bf16*)alloc(wBytes);
    __bf16* Wqt = (__bf16*)alloc(wBytes);
    __bf16* Wvt = (__bf16*)alloc(wBytes);
    __bf16* Wpt = (__bf16*)alloc(wBytes);
    __bf16* Kb  = (__bf16*)alloc(actBytes);
    __bf16* Qb  = (__bf16*)alloc(actBytes);
    __bf16* Vb  = (__bf16*)alloc(actBytes);
    __bf16* Vtb = (__bf16*)alloc(actBytes);
    __bf16* Ab  = (__bf16*)alloc(actBytes);

    const int nAct = BT * Cc;
    cvt_f32_bf16<<<(nAct + 255) / 256, 256, 0, stream>>>(x, xb, nAct);
    const int nW = Cc * Cc;
    transpose_cvt_w<<<(nW + 255) / 256, 256, 0, stream>>>(Wk, Wkt, Cc, Cc);
    transpose_cvt_w<<<(nW + 255) / 256, 256, 0, stream>>>(Wq, Wqt, Cc, Cc);
    transpose_cvt_w<<<(nW + 255) / 256, 256, 0, stream>>>(Wv, Wvt, Cc, Cc);
    transpose_cvt_w<<<(nW + 255) / 256, 256, 0, stream>>>(Wp, Wpt, Cc, Cc);

    dim3 gg(BT / 32, Cc / 256), gb(128);
    gemm_bf16<<<gg, gb, 0, stream>>>(xb, Wkt, bk, Kb, nullptr, BT, Cc, Cc);
    gemm_bf16<<<gg, gb, 0, stream>>>(xb, Wqt, bq, Qb, nullptr, BT, Cc, Cc);
    gemm_bf16<<<gg, gb, 0, stream>>>(xb, Wvt, bv, Vb, nullptr, BT, Cc, Cc);

    const int nRot = BT * (Cc / 2);
    rope_rotate<<<(nRot + 255) / 256, 256, 0, stream>>>(Kb, nRot);
    rope_rotate<<<(nRot + 255) / 256, 256, 0, stream>>>(Qb, nRot);
    transpose_v<<<(nAct + 255) / 256, 256, 0, stream>>>(Vb, Vtb, nAct);

    attn_kernel<<<dim3(Tt / 16, Bb), 256, 16 * Tt * 2, stream>>>(Kb, Qb, Vtb, Ab);

    gemm_bf16<<<gg, gb, 0, stream>>>(Ab, Wpt, bp, nullptr, out, BT, Cc, Cc);
}
